// ViT_47656957116789
// MI455X (gfx1250) — matrix operand order repaired and checked by reading, not compiled, not run on hardware
//
#include <hip/hip_runtime.h>
#include <hip/hip_bf16.h>

// ---------------------------------------------------------------------------
// ViT forward: patch-embed -> QKV -> flash-attention -> output projection
// All GEMMs use v_wmma_f32_16x16x32_f16 (CDNA5 / gfx1250, wave32).
// ---------------------------------------------------------------------------

typedef __attribute__((ext_vector_type(16))) _Float16 v16h;
typedef __attribute__((ext_vector_type(8)))  _Float16 v8h;
typedef __attribute__((ext_vector_type(8)))  float    v8f;

#define IMG     256
#define GRIDP   64          // 256/4 patches per side
#define NTOK    4096        // 64*64 tokens per batch image
#define BATCH   8
#define TOT_TOK (BATCH * NTOK)   // 32768
#define EMB     768
#define HD      96          // k/q/v head dim

// Assemble a v16h WMMA operand from two 16-byte chunks (global or LDS).
__device__ __forceinline__ v16h ld16(const _Float16* p0, const _Float16* p1) {
    union { v16h v; v8h h[2]; } u;
    u.h[0] = *(const v8h*)p0;
    u.h[1] = *(const v8h*)p1;
    return u.v;
}

__device__ __forceinline__ v8f wmma16(v16h a, v16h b, v8f c) {
    return __builtin_amdgcn_wmma_f32_16x16x32_f16(
        /*neg_a=*/false, a, /*neg_b=*/false, b,
        /*c_mod=*/(short)0, c, /*reuse_a=*/false, /*reuse_b=*/false);
}

// ---------------------------------------------------------------------------
// Kernel 0: fp32 -> fp16 weight conversion (runs once per launch; tiny)
// ---------------------------------------------------------------------------
__global__ void vit_f32_to_f16(const float* __restrict__ src,
                               _Float16* __restrict__ dst, int n) {
    int i = blockIdx.x * blockDim.x + threadIdx.x;
    if (i < n) dst[i] = (_Float16)src[i];
}

// ---------------------------------------------------------------------------
// Kernel 1: patch embedding. One block per token; 48-element dot per output.
// tokens_h[t][e] = conv_b[e] + sum_{c,i,j} x[b,c,4ph+i,4pw+j] * conv_w[e,c,i,j]
// ---------------------------------------------------------------------------
__global__ void vit_patch_embed(const float* __restrict__ x,
                                const float* __restrict__ conv_w,
                                const float* __restrict__ conv_b,
                                _Float16* __restrict__ tokens_h) {
    __shared__ float patch[48];
    const int t  = blockIdx.x;          // global token 0..32767
    const int b  = t >> 12;             // /4096
    const int p  = t & (NTOK - 1);
    const int ph = p >> 6, pw = p & 63;
    const int tid = threadIdx.x;        // 256 threads

    if (tid < 48) {
        const int c = tid >> 4, rem = tid & 15;
        const int ii = rem >> 2, jj = rem & 3;
        patch[tid] = x[(((size_t)(b * 3 + c) * IMG) + ph * 4 + ii) * IMG + pw * 4 + jj];
    }
    __syncthreads();

    #pragma unroll
    for (int eo = 0; eo < 3; ++eo) {
        const int e = tid + eo * 256;
        float acc = conv_b[e];
        const float* w = conv_w + (size_t)e * 48;
        #pragma unroll
        for (int kk = 0; kk < 48; ++kk) acc += patch[kk] * w[kk];
        tokens_h[(size_t)t * EMB + e] = (_Float16)acc;
    }
}

// ---------------------------------------------------------------------------
// Kernel 2: Q/K/V projection. grid = (TOT_TOK/16, 18); 1 wave per block.
// blockIdx.y: sel = y/6 (0=q,1=k,2=v), column tile = (y%6)*16.
// V is stored transposed [B, 96, N] so the PV GEMM's B operand is contiguous.
// ---------------------------------------------------------------------------
__global__ void vit_qkv(const _Float16* __restrict__ tokens_h,
                        const _Float16* __restrict__ kw_h,
                        const _Float16* __restrict__ vw_h,
                        const _Float16* __restrict__ qw_h,
                        const float* __restrict__ k_b,
                        const float* __restrict__ v_b,
                        const float* __restrict__ q_b,
                        _Float16* __restrict__ q_h,
                        _Float16* __restrict__ k_h,
                        _Float16* __restrict__ vT_h) {
    const int l = threadIdx.x;
    const int lane16 = l & 15, h = l >> 4;
    const int m0  = blockIdx.x * 16;
    const int sel = blockIdx.y / 6;
    const int c0  = (blockIdx.y % 6) * 16;

    const _Float16* w    = (sel == 0) ? qw_h : (sel == 1) ? kw_h : vw_h;
    const float*    bias = (sel == 0) ? q_b  : (sel == 1) ? k_b  : v_b;

    const _Float16* arow = tokens_h + (size_t)(m0 + lane16) * EMB;   // A row
    const _Float16* brow = w + (size_t)(c0 + lane16) * EMB;          // B column

    v8f acc = {};
    #pragma unroll 4
    for (int ks = 0; ks < EMB / 32; ++ks) {
        const int k0 = ks * 32;
        v16h a  = ld16(arow + k0 + h * 8,  arow + k0 + 16 + h * 8);
        v16h bm = ld16(brow + k0 + h * 8,  brow + k0 + 16 + h * 8);
        acc = wmma16(a, bm, acc);
    }

    const float bv = bias[c0 + lane16];
    #pragma unroll
    for (int r = 0; r < 8; ++r) {
        const float val = acc[r] + bv;
        const int tk = m0 + h * 8 + r;       // token row of D element r
        const int d  = c0 + lane16;          // output dim (D column)
        if (sel == 2) {                      // V: store transposed [B,96,N]
            const int bb = tk >> 12, n = tk & (NTOK - 1);
            vT_h[((size_t)bb * HD + d) * NTOK + n] = (_Float16)val;
        } else {
            _Float16* dst = (sel == 0) ? q_h : k_h;
            dst[(size_t)tk * HD + d] = (_Float16)val;
        }
    }
}

// ---------------------------------------------------------------------------
// Kernel 3: flash attention. grid = (N/16, B); 1 wave per block.
// Wave owns a 16-query tile; streams 32-key blocks with online softmax;
// O (16x96) + row stats live in VGPRs; P reshaped D->A layout via 1KB LDS.
// ---------------------------------------------------------------------------
__global__ void vit_flash_attn(const _Float16* __restrict__ q_h,
                               const _Float16* __restrict__ k_h,
                               const _Float16* __restrict__ vT_h,
                               _Float16* __restrict__ attn_h) {
    __shared__ _Float16 lds_p[16 * 32];   // 16 rows x 32 key cols, fp16

    const int l = threadIdx.x;
    const int lane16 = l & 15, h = l >> 4;
    const int b  = blockIdx.y;
    const int q0 = blockIdx.x * 16;
    const float scale = 0.10206207262f;   // 1/sqrt(96)

    const _Float16* qb  = q_h  + (size_t)b * NTOK * HD;
    const _Float16* kbp = k_h  + (size_t)b * NTOK * HD;
    const _Float16* vTb = vT_h + (size_t)b * HD * NTOK;

    // Q tile resident in registers: 3 A-operands (K=32 each, total K=96)
    v16h qa[3];
    const _Float16* qrow = qb + (size_t)(q0 + lane16) * HD;
    #pragma unroll
    for (int ks = 0; ks < 3; ++ks)
        qa[ks] = ld16(qrow + ks * 32 + h * 8, qrow + ks * 32 + 16 + h * 8);

    v8f o[6];
    v8f zero = {};
    #pragma unroll
    for (int ct = 0; ct < 6; ++ct) o[ct] = zero;
    float mrow[8], lrow[8];
    #pragma unroll
    for (int r = 0; r < 8; ++r) { mrow[r] = -3.0e38f; lrow[r] = 0.0f; }

    for (int kb0 = 0; kb0 < NTOK; kb0 += 32) {
        // ---- scores: S[16x32] = Q @ K^T (two 16-key column tiles) ----
        v8f s0 = zero, s1 = zero;
        {
            const _Float16* kr0 = kbp + (size_t)(kb0 + lane16) * HD;
            const _Float16* kr1 = kbp + (size_t)(kb0 + 16 + lane16) * HD;
            #pragma unroll
            for (int ks = 0; ks < 3; ++ks) {
                v16h b0 = ld16(kr0 + ks * 32 + h * 8, kr0 + ks * 32 + 16 + h * 8);
                s0 = wmma16(qa[ks], b0, s0);
                v16h b1 = ld16(kr1 + ks * 32 + h * 8, kr1 + ks * 32 + 16 + h * 8);
                s1 = wmma16(qa[ks], b1, s1);
            }
        }

        // ---- online softmax (row = query). D-layout: lane holds key col
        // lane16 for rows M = h*8+r; row reduce = xor-shuffle over 16 lanes.
        float p0[8], p1[8];
        #pragma unroll
        for (int r = 0; r < 8; ++r) {
            const float x0 = s0[r] * scale;
            const float x1 = s1[r] * scale;
            float tm = fmaxf(x0, x1);
            tm = fmaxf(tm, __shfl_xor(tm, 1));
            tm = fmaxf(tm, __shfl_xor(tm, 2));
            tm = fmaxf(tm, __shfl_xor(tm, 4));
            tm = fmaxf(tm, __shfl_xor(tm, 8));
            const float nm  = fmaxf(mrow[r], tm);
            const float fac = __expf(mrow[r] - nm);
            p0[r] = __expf(x0 - nm);
            p1[r] = __expf(x1 - nm);
            float ps = p0[r] + p1[r];
            ps += __shfl_xor(ps, 1);
            ps += __shfl_xor(ps, 2);
            ps += __shfl_xor(ps, 4);
            ps += __shfl_xor(ps, 8);
            lrow[r] = lrow[r] * fac + ps;
            mrow[r] = nm;
            #pragma unroll
            for (int ct = 0; ct < 6; ++ct) o[ct][r] *= fac;
        }

        // ---- reshape P: D-layout -> A-layout through LDS ----
        #pragma unroll
        for (int r = 0; r < 8; ++r) {
            lds_p[(h * 8 + r) * 32 + lane16]      = (_Float16)p0[r];
            lds_p[(h * 8 + r) * 32 + 16 + lane16] = (_Float16)p1[r];
        }
        __syncthreads();
        v16h pa = ld16(&lds_p[lane16 * 32 + h * 8],
                       &lds_p[lane16 * 32 + 16 + h * 8]);
        __syncthreads();

        // ---- O += P @ V  (V transposed -> contiguous B operand) ----
        #pragma unroll
        for (int ct = 0; ct < 6; ++ct) {
            const _Float16* vrow = vTb + (size_t)(ct * 16 + lane16) * NTOK + kb0;
            v16h bv = ld16(vrow + h * 8, vrow + 16 + h * 8);
            o[ct] = wmma16(pa, bv, o[ct]);
        }
    }

    // ---- finalize: O /= rowsum; store fp16 [TOT_TOK, 96] ----
    #pragma unroll
    for (int r = 0; r < 8; ++r) {
        const float inv = 1.0f / lrow[r];
        const size_t tk = (size_t)b * NTOK + q0 + h * 8 + r;
        #pragma unroll
        for (int ct = 0; ct < 6; ++ct)
            attn_h[tk * HD + ct * 16 + lane16] = (_Float16)(o[ct][r] * inv);
    }
}

// ---------------------------------------------------------------------------
// Kernel 4: output projection. grid = (TOT_TOK/16, EMB/16); 1 wave per block.
// out[t][e] = attn[t][:] . proj_w[e][:] + proj_b[e]   (K=96, fp32 out)
// ---------------------------------------------------------------------------
__global__ void vit_proj(const _Float16* __restrict__ attn_h,
                         const _Float16* __restrict__ pw_h,
                         const float* __restrict__ proj_b,
                         float* __restrict__ out) {
    const int l = threadIdx.x;
    const int lane16 = l & 15, h = l >> 4;
    const int m0 = blockIdx.x * 16;
    const int e0 = blockIdx.y * 16;

    const _Float16* arow = attn_h + (size_t)(m0 + lane16) * HD;
    const _Float16* brow = pw_h + (size_t)(e0 + lane16) * HD;

    v8f acc = {};
    #pragma unroll
    for (int ks = 0; ks < 3; ++ks) {
        v16h a  = ld16(arow + ks * 32 + h * 8,  arow + ks * 32 + 16 + h * 8);
        v16h bm = ld16(brow + ks * 32 + h * 8,  brow + ks * 32 + 16 + h * 8);
        acc = wmma16(a, bm, acc);
    }

    const float bias = proj_b[e0 + lane16];
    #pragma unroll
    for (int r = 0; r < 8; ++r)
        out[(size_t)(m0 + h * 8 + r) * EMB + e0 + lane16] = acc[r] + bias;
}

// ---------------------------------------------------------------------------
// Launch
// ---------------------------------------------------------------------------
extern "C" void kernel_launch(void* const* d_in, const int* in_sizes, int n_in,
                              void* d_out, int out_size, void* d_ws, size_t ws_size,
                              hipStream_t stream) {
    const float* x      = (const float*)d_in[0];
    const float* conv_w = (const float*)d_in[1];
    const float* conv_b = (const float*)d_in[2];
    const float* k_w    = (const float*)d_in[3];
    const float* k_b    = (const float*)d_in[4];
    const float* v_w    = (const float*)d_in[5];
    const float* v_b    = (const float*)d_in[6];
    const float* q_w    = (const float*)d_in[7];
    const float* q_b    = (const float*)d_in[8];
    const float* proj_w = (const float*)d_in[9];
    const float* proj_b = (const float*)d_in[10];
    float* out = (float*)d_out;

    // Workspace layout (fp16). Total ~76 MB.
    _Float16* ws = (_Float16*)d_ws;
    _Float16* tokens_h = ws;  ws += (size_t)TOT_TOK * EMB;   // 48 MB
    _Float16* q_h      = ws;  ws += (size_t)TOT_TOK * HD;    // 6 MB
    _Float16* k_h      = ws;  ws += (size_t)TOT_TOK * HD;    // 6 MB
    _Float16* vT_h     = ws;  ws += (size_t)TOT_TOK * HD;    // 6 MB
    _Float16* attn_h   = ws;  ws += (size_t)TOT_TOK * HD;    // 6 MB
    _Float16* kw_h     = ws;  ws += (size_t)HD * EMB;
    _Float16* vw_h     = ws;  ws += (size_t)HD * EMB;
    _Float16* qw_h     = ws;  ws += (size_t)HD * EMB;
    _Float16* pw_h     = ws;  ws += (size_t)EMB * HD;

    const int nw = HD * EMB;  // 73728 (same count for proj_w)
    vit_f32_to_f16<<<(nw + 255) / 256, 256, 0, stream>>>(k_w, kw_h, nw);
    vit_f32_to_f16<<<(nw + 255) / 256, 256, 0, stream>>>(v_w, vw_h, nw);
    vit_f32_to_f16<<<(nw + 255) / 256, 256, 0, stream>>>(q_w, qw_h, nw);
    vit_f32_to_f16<<<(nw + 255) / 256, 256, 0, stream>>>(proj_w, pw_h, nw);

    vit_patch_embed<<<TOT_TOK, 256, 0, stream>>>(x, conv_w, conv_b, tokens_h);

    vit_qkv<<<dim3(TOT_TOK / 16, 18), 32, 0, stream>>>(
        tokens_h, kw_h, vw_h, qw_h, k_b, v_b, q_b, q_h, k_h, vT_h);

    vit_flash_attn<<<dim3(NTOK / 16, BATCH), 32, 0, stream>>>(
        q_h, k_h, vT_h, attn_h);

    vit_proj<<<dim3(TOT_TOK / 16, EMB / 16), 32, 0, stream>>>(
        attn_h, pw_h, proj_b, out);
}
